// LSTMchain_3410204033763
// MI455X (gfx1250) — matrix operand order repaired and checked by reading, not compiled, not run on hardware
//
#include <hip/hip_runtime.h>

typedef __bf16 bf16_t;
typedef __attribute__((ext_vector_type(16))) __bf16 v16bf;
typedef __attribute__((ext_vector_type(8)))  float  v8f;

#define Hh   256   // hidden
#define Ss   512   // seq len
#define Bb   256   // batch
#define Vv   128   // vocab
#define K2   512   // 2H (concat K)
#define N4   1024  // 4H (fused gate cols)
#define MT   2     // M-tiles per workgroup (M = 32 batch rows)

union FragBF { v16bf v; uint4 q[2]; };

// branchless fast sigmoid: v_exp_f32 + v_rcp_f32, no EXEC divergence, no div chain
__device__ __forceinline__ float sigmoid_(float x) {
    return __builtin_amdgcn_rcpf(1.0f + __expf(-x));
}

// branchless, overflow-safe fast tanh: t = exp(-2|x|) in (0,1]
__device__ __forceinline__ float tanh_(float x) {
    float ax = __builtin_fabsf(x);
    float t  = __expf(-2.0f * ax);
    float r  = (1.0f - t) * __builtin_amdgcn_rcpf(1.0f + t);
    return __builtin_copysignf(r, x);
}

__device__ __forceinline__ v8f splat8(float x) {
    v8f r;
#pragma unroll
    for (int i = 0; i < 8; ++i) r[i] = x;
    return r;
}

// ---- prep kernel 1: emb f32 -> bf16 ----
__global__ __launch_bounds__(256) void cvt_emb_kernel(const float* __restrict__ src,
                                                      bf16_t* __restrict__ dst, int n) {
    int i = blockIdx.x * blockDim.x + threadIdx.x;
    if (i < n) dst[i] = (bf16_t)src[i];
}

// ---- prep kernel 2: W [K2, N4] f32 -> fragment-major bf16 ----
// Wfrag index = ((ntile*16 + k) * 32 + lane) * 16 + e
//   ntile = 0..63 (N-tile of 16 cols), k = 0..15 (K-chunk of 32),
//   lane 0..31, e 0..15 : B operand for v_wmma_f32_16x16x32_bf16 in this file's k order (same as A)
//   (lanes 0-15 hold K = kb+0..15 of col lane; lanes 16-31 hold K = kb+16..31)
__global__ __launch_bounds__(256) void pack_w_kernel(const float* __restrict__ W,
                                                     bf16_t* __restrict__ Wfrag) {
    int i = blockIdx.x * blockDim.x + threadIdx.x;  // over N4*K2, coalesced writes
    int e     = i & 15;
    int lane  = (i >> 4) & 31;
    int k     = (i >> 9) & 15;
    int ntile = i >> 13;                            // 0..63
    int n  = ntile * 16 + (lane & 15);
    int kk = k * 32 + (lane >> 4) * 16 + e;
    Wfrag[i] = (bf16_t)W[kk * N4 + n];
}

// ---- main recurrent kernel ----
// grid: B/32 workgroups; block: 512 threads = 16 waves (wave32).
// Each workgroup: 32 batch rows (2 WMMA M-tiles) for all S steps.
// Wave w owns H-columns [16w, 16w+16): N-tile (g*16 + w) for each gate g,
// so the i/f/o/u fragments for a column live in one wave and the cell state
// stays in VGPRs for all 512 steps.
__global__ __launch_bounds__(512) void lstm_seq_kernel(
    const int*    __restrict__ tokens,  // [B,S]
    const bf16_t* __restrict__ embb,    // [V,H] bf16
    const bf16_t* __restrict__ Wfrag,   // fragment-major W, bf16
    const float*  __restrict__ bias,    // [N4]
    float*        __restrict__ out)     // [B,S,H] f32
{
    __shared__ __align__(16) bf16_t hx[32][K2];  // concat [h | x_t], 32 KB

    const int tid   = threadIdx.x;
    const int lane  = tid & 31;
    const int wave  = tid >> 5;        // 0..15
    const int lrow  = lane & 15;       // 0..15
    const int lhi   = lane >> 4;       // 0/1
    const int bbase = blockIdx.x * 32;

    // cooperative x-gather: thread handles row xm, 16 cols starting at xc
    const int xm = tid >> 4;           // 0..31
    const int xc = (tid & 15) * 16;    // 0..240

    // init h = 0 in LDS
    {
        uint4 z4 = make_uint4(0u, 0u, 0u, 0u);
        uint4* p = reinterpret_cast<uint4*>(&hx[xm][xc]);
        p[0] = z4; p[1] = z4;
    }
    // gather x_0
    {
        int tok = tokens[(bbase + xm) * Ss + 0];
        const uint4* s = reinterpret_cast<const uint4*>(&embb[tok * Hh + xc]);
        uint4* d = reinterpret_cast<uint4*>(&hx[xm][Hh + xc]);
        d[0] = s[0]; d[1] = s[1];
    }

    // per-lane bias for the 4 gate tiles this wave owns
    float bv[4];
#pragma unroll
    for (int g = 0; g < 4; ++g) bv[g] = bias[256 * g + 16 * wave + lrow];

    // per-gate fragment bases into Wfrag (per-lane contiguous 32B slabs)
    const bf16_t* wbase[4];
#pragma unroll
    for (int g = 0; g < 4; ++g)
        wbase[g] = Wfrag + ((size_t)(g * 16 + wave) * 16) * 512 + lane * 16;

    // persistent cell state in accumulator-fragment layout
    v8f cst[MT];
#pragma unroll
    for (int mt = 0; mt < MT; ++mt) cst[mt] = splat8(0.0f);

#pragma unroll 1
    for (int t = 0; t < Ss; ++t) {
        __syncthreads();   // hx holds h_t (cols 0..H) and x_t (cols H..2H)

        v8f acc[MT][4];
#pragma unroll
        for (int mt = 0; mt < MT; ++mt)
#pragma unroll
            for (int g = 0; g < 4; ++g) acc[mt][g] = splat8(bv[g]);

        // K loop: kept rolled so B-fragment loads stream from L2 each step
        // (prevents LICM hoist + scratch spill of the whole W tile)
#pragma unroll 1
        for (int k = 0; k < 16; ++k) {
            const int kb = k * 32;
            FragBF a[MT];
#pragma unroll
            for (int mt = 0; mt < MT; ++mt) {
                const bf16_t* ap = &hx[mt * 16 + lrow][kb + lhi * 16];
                a[mt].q[0] = *reinterpret_cast<const uint4*>(ap);
                a[mt].q[1] = *reinterpret_cast<const uint4*>(ap + 8);
            }
#pragma unroll
            for (int g = 0; g < 4; ++g) {
                FragBF bf;
                const uint4* bp =
                    reinterpret_cast<const uint4*>(wbase[g] + (size_t)k * 512);
                bf.q[0] = bp[0];
                bf.q[1] = bp[1];
#pragma unroll
                for (int mt = 0; mt < MT; ++mt)
                    acc[mt][g] = __builtin_amdgcn_wmma_f32_16x16x32_bf16(
                        false, a[mt].v, false, bf.v, (short)0, acc[mt][g],
                        false, false);
            }
        }

        __syncthreads();   // all reads of hx done before overwrite

        // gates + state update entirely in this wave's fragments (branchless)
        const int hcol = 16 * wave + lrow;
#pragma unroll
        for (int mt = 0; mt < MT; ++mt) {
#pragma unroll
            for (int r = 0; r < 8; ++r) {
                float iv = sigmoid_(acc[mt][0][r]);
                float fv = sigmoid_(acc[mt][1][r]);
                float ov = sigmoid_(acc[mt][2][r]);
                float uv = tanh_(acc[mt][3][r]);
                float cv = iv * uv + fv * cst[mt][r];
                cst[mt][r] = cv;
                float hv = ov * tanh_(cv);
                const int m = mt * 16 + lhi * 8 + r;
                out[((size_t)(bbase + m) * Ss + t) * Hh + hcol] = hv;
                hx[m][hcol] = (bf16_t)hv;   // new h for next step
            }
        }

        // cooperative gather of x_{t+1}
        if (t + 1 < Ss) {
            int tok = tokens[(bbase + xm) * Ss + (t + 1)];
            const uint4* s = reinterpret_cast<const uint4*>(&embb[tok * Hh + xc]);
            uint4* d = reinterpret_cast<uint4*>(&hx[xm][Hh + xc]);
            d[0] = s[0]; d[1] = s[1];
        }
    }
}

extern "C" void kernel_launch(void* const* d_in, const int* in_sizes, int n_in,
                              void* d_out, int out_size, void* d_ws, size_t ws_size,
                              hipStream_t stream) {
    const int*   tokens = (const int*)d_in[0];    // [B,S]
    const float* emb    = (const float*)d_in[1];  // [V,H]
    const float* W      = (const float*)d_in[2];  // [2H,4H]
    const float* b      = (const float*)d_in[3];  // [4H]
    float*       out    = (float*)d_out;          // [B,S,H]

    bf16_t* embb  = (bf16_t*)d_ws;                        // 64 KB
    bf16_t* Wfrag = (bf16_t*)((char*)d_ws + 64 * 1024);   // 1 MB

    cvt_emb_kernel<<<(Vv * Hh + 255) / 256, 256, 0, stream>>>(emb, embb, Vv * Hh);
    pack_w_kernel<<<(N4 * K2) / 256, 256, 0, stream>>>(W, Wfrag);
    lstm_seq_kernel<<<Bb / 32, 512, 0, stream>>>(tokens, embb, Wfrag, b, out);
}
